// ModalityUntiedAttention_16561393893892
// MI455X (gfx1250) — compile-verified, matrix operand order checked by reading, not run on hardware
//
#include <hip/hip_runtime.h>

// ModalityUntiedAttention for MI455X (gfx1250, wave32).
// bf16 WMMA (v_wmma_f32_16x16x32_bf16) everywhere; flash-attention (no
// 256MB score tensor); block-cooperative double-buffered async K/V staging
// into LDS (GLOBAL_LOAD_ASYNC_TO_LDS_B128 / ASYNCcnt) -- all tensors are
// L2-resident (192MB), so the kernel is matrix-throughput bound.

#define N_TOK 2048
#define DIM   1024
#define NHEAD 16
#define HDIM  64
#define NMOD  2

typedef __attribute__((ext_vector_type(16))) __bf16 bf16x16;
typedef __attribute__((ext_vector_type(8)))  __bf16 bf16x8;
typedef __attribute__((ext_vector_type(8)))  float  f32x8;

// The async-load builtin is declared with gcc-vector int4 pointers in
// address spaces 1 (global, printed as __device__) and 3 (LDS).
typedef int v4i_gcc __attribute__((vector_size(16)));
typedef __attribute__((address_space(1))) v4i_gcc as1_v4i;
typedef __attribute__((address_space(3))) v4i_gcc as3_v4i;

static __device__ __forceinline__ f32x8 wmma_bf16(bf16x16 a, bf16x16 b, f32x8 c) {
  // D = A(16x32 bf16) * B(32x16 bf16) + C(16x16 f32)
  return __builtin_amdgcn_wmma_f32_16x16x32_bf16(false, a, false, b, (short)0, c,
                                                 false, false);
}

// 16-byte global -> LDS async copy (ASYNCcnt-tracked). Fallback: direct copy.
static __device__ __forceinline__ void async_copy16(const __bf16* g, __bf16* l) {
#if __has_builtin(__builtin_amdgcn_global_load_async_to_lds_b128)
  // flat->as(3) is a low-32-bit truncation on amdgcn; flat->as(1) is identity.
  __builtin_amdgcn_global_load_async_to_lds_b128(
      (as1_v4i*)(size_t)g, (as3_v4i*)(unsigned)(size_t)l, 0, 0);
#else
  *(bf16x8*)l = *(const bf16x8*)g;
#endif
}

static __device__ __forceinline__ void wait_async0() {
#if __has_builtin(__builtin_amdgcn_s_wait_asynccnt)
  __builtin_amdgcn_s_wait_asynccnt(0);
#else
  asm volatile("s_wait_asynccnt 0x0" ::: "memory");
#endif
}

// A/B fragment loader from a row-major bf16 matrix (global or LDS). Per ISA
// 7.12.2, lane (16*hi + r) holds line r; element e maps to K = 8*hi+e (e<8)
// or 16+8*hi+(e-8) (e>=8): two contiguous 16-byte runs per lane.
static __device__ __forceinline__ bf16x16
load_frag_bf16(const __bf16* base, int stride, int row, int k0, int hi) {
  const __bf16* p0 = base + (size_t)row * stride + k0 + hi * 8;
  bf16x8 lo = *(const bf16x8*)(p0);
  bf16x8 hh = *(const bf16x8*)(p0 + 16);
  bf16x16 f;
#pragma unroll
  for (int e = 0; e < 8; ++e) { f[e] = lo[e]; f[e + 8] = hh[e]; }
  return f;
}

// ---------------- kernel 0: modality id per token --------------------------
__global__ __launch_bounds__(256) void mods_kernel(
    const unsigned char* __restrict__ masks, int* __restrict__ mod) {
  int n = blockIdx.x * blockDim.x + threadIdx.x;
  if (n >= N_TOK) return;
  int sel = 0;
#pragma unroll
  for (int m = 0; m < NMOD; ++m)
    if (masks[(size_t)m * N_TOK + n]) { sel = m; break; }  // argmax over bools
  mod[n] = sel;
}

// ------------- kernel 1a: weight transpose + bf16 convert ------------------
// in:  w[d][o] fp32 ; out: wt[o][d] bf16  (B-fragment loads become contiguous)
__global__ __launch_bounds__(256) void wt_kernel(
    const float* __restrict__ w, __bf16* __restrict__ wt, int din, int dout) {
  size_t idx = (size_t)blockIdx.x * blockDim.x + threadIdx.x;
  if (idx >= (size_t)din * dout) return;
  int d = (int)(idx / dout), o = (int)(idx % dout);
  wt[(size_t)o * din + d] = (__bf16)w[idx];
}

// ------------- kernel 1b: plain fp32 -> bf16 convert (for x) ---------------
__global__ __launch_bounds__(256) void cvt_kernel(
    const float* __restrict__ in, __bf16* __restrict__ outb, int n) {
  int i = blockIdx.x * blockDim.x + threadIdx.x;
  if (i < n) outb[i] = (__bf16)in[i];
}

// ------------- kernel 2: fused QKV expert projection -----------------------
// wt layout: [proj(3)][mod(2)][out DIM][in DIM] bf16. Each wave: one
// 16(token) x 16(out) tile accumulated with BOTH experts' weights (M=2),
// per-row select in the epilogue. V stored transposed for the PV B-layout.
__global__ __launch_bounds__(256) void qkv_proj_kernel(
    const __bf16* __restrict__ xb, const __bf16* __restrict__ wt,
    const int* __restrict__ mod, __bf16* __restrict__ qb,
    __bf16* __restrict__ kb, __bf16* __restrict__ vT) {
  const int lane = threadIdx.x & 31, wave = threadIdx.x >> 5;
  const int hi = lane >> 4, lm = lane & 15;
  const int t0 = blockIdx.x * 16;             // token tile
  const int ct = blockIdx.y * 8 + wave;       // 0..191
  const int p  = ct >> 6;                     // 0=q,1=k,2=v
  const int c0 = (ct & 63) * 16;              // output col base
  const __bf16* w0 = wt + ((size_t)p * NMOD + 0) * DIM * DIM;
  const __bf16* w1 = w0 + (size_t)DIM * DIM;

  f32x8 acc0 = {}, acc1 = {};
  for (int k0 = 0; k0 < DIM; k0 += 32) {
    bf16x16 a  = load_frag_bf16(xb, DIM, t0 + lm, k0, hi);
    bf16x16 b0 = load_frag_bf16(w0, DIM, c0 + lm, k0, hi);
    bf16x16 b1 = load_frag_bf16(w1, DIM, c0 + lm, k0, hi);
    acc0 = wmma_bf16(a, b0, acc0);
    acc1 = wmma_bf16(a, b1, acc1);
  }
#pragma unroll
  for (int r = 0; r < 8; ++r) {               // C/D: row r+8*hi, col lm
    int row = t0 + r + 8 * hi;
    int col = c0 + lm;
    float v = mod[row] ? acc1[r] : acc0[r];
    __bf16 bv = (__bf16)v;
    if (p == 0)      qb[(size_t)row * DIM + col] = bv;
    else if (p == 1) kb[(size_t)row * DIM + col] = bv;
    else             vT[(size_t)col * N_TOK + row] = bv;   // V transposed
  }
}

// ------------- kernel 3: flash attention -----------------------------------
// One block = one head x 128 queries (8 waves x 16-query tiles). All waves
// share the 32-key K/V chunk, staged into LDS with double-buffered async
// copies overlapped with the WMMA + online-softmax compute.
__global__ __launch_bounds__(256) void attn_kernel(
    const __bf16* __restrict__ qb, const __bf16* __restrict__ kb,
    const __bf16* __restrict__ vT, __bf16* __restrict__ ob) {
  __shared__ __bf16 kt[2][32 * 64];           // [buf][key][hd]   2 x 4KB
  __shared__ __bf16 vt[2][64 * 32];           // [buf][hd][key]   2 x 4KB
  __shared__ __bf16 pbuf[8][16 * 32];         // per-wave P staging, 8KB
  const int tid = threadIdx.x;
  const int lane = tid & 31, wave = tid >> 5;
  const int hi = lane >> 4, lm = lane & 15;
  const int h  = blockIdx.x;
  const int q0 = (blockIdx.y * 8 + wave) * 16;
  const __bf16* Kg = kb + h * HDIM;                     // row stride DIM
  const __bf16* Vg = vT + (size_t)h * HDIM * N_TOK;     // [hd][token]
  __bf16* pl = &pbuf[wave][0];
  const float scale = 0.125f;                 // HD^-0.5

  bf16x16 qa0 = load_frag_bf16(qb + h * HDIM, DIM, q0 + lm, 0, hi);
  bf16x16 qa1 = load_frag_bf16(qb + h * HDIM, DIM, q0 + lm, 32, hi);

  // cooperative tile fetch: 16B per thread per tensor (256 thr x 16B = 4KB)
  const int kr = tid >> 3, ks = (tid & 7) * 8;   // K: [key 32][hd 64]
  const int vr = tid >> 2, vs = (tid & 3) * 8;   // V: [hd 64][key 32]

  // preload chunk 0 into buffer 0
  async_copy16(Kg + (size_t)kr * DIM + ks, &kt[0][kr * 64 + ks]);
  async_copy16(Vg + (size_t)vr * N_TOK + vs, &vt[0][vr * 32 + vs]);

  f32x8 o0 = {}, o1 = {}, o2 = {}, o3 = {};
  float mrow[8], lrow[8];
#pragma unroll
  for (int r = 0; r < 8; ++r) { mrow[r] = -1e30f; lrow[r] = 0.f; }

  for (int kk = 0; kk < N_TOK; kk += 32) {
    const int buf = (kk >> 5) & 1;
    wait_async0();                            // this wave's LDS fills landed
    __syncthreads();                          // chunk published, prev reads done
    if (kk + 32 < N_TOK) {                    // prefetch next chunk
      async_copy16(Kg + (size_t)(kk + 32 + kr) * DIM + ks,
                   &kt[buf ^ 1][kr * 64 + ks]);
      async_copy16(Vg + (size_t)vr * N_TOK + (kk + 32) + vs,
                   &vt[buf ^ 1][vr * 32 + vs]);
    }
    const __bf16* kp = &kt[buf][0];
    const __bf16* vp = &vt[buf][0];

    // S = Q(16x64) * K^T : two 16-key tiles, K-dim chained (64 = 2x32)
    f32x8 s0 = {}, s1 = {};
    s0 = wmma_bf16(qa0, load_frag_bf16(kp, 64, lm, 0, hi), s0);
    s0 = wmma_bf16(qa1, load_frag_bf16(kp, 64, lm, 32, hi), s0);
    s1 = wmma_bf16(qa0, load_frag_bf16(kp, 64, 16 + lm, 0, hi), s1);
    s1 = wmma_bf16(qa1, load_frag_bf16(kp, 64, 16 + lm, 32, hi), s1);

    // online softmax; row = r + 8*hi, 16-lane reductions stay inside halves
#pragma unroll
    for (int r = 0; r < 8; ++r) {
      float a = s0[r] * scale, b = s1[r] * scale;
      float mx = fmaxf(a, b);
#pragma unroll
      for (int m = 1; m < 16; m <<= 1) mx = fmaxf(mx, __shfl_xor(mx, m, 32));
      float mnew  = fmaxf(mrow[r], mx);
      float alpha = __expf(mrow[r] - mnew);
      float pa = __expf(a - mnew), pb = __expf(b - mnew);
      float ps = pa + pb;
#pragma unroll
      for (int m = 1; m < 16; m <<= 1) ps += __shfl_xor(ps, m, 32);
      lrow[r] = lrow[r] * alpha + ps;
      mrow[r] = mnew;
      o0[r] *= alpha; o1[r] *= alpha; o2[r] *= alpha; o3[r] *= alpha;
      int row = r + 8 * hi;                   // C-layout -> LDS row-major P
      pl[row * 32 + lm]      = (__bf16)pa;
      pl[row * 32 + 16 + lm] = (__bf16)pb;
    }
    asm volatile("s_wait_dscnt 0x0" ::: "memory");

    bf16x16 pf = load_frag_bf16(pl, 32, lm, 0, hi);     // P as A-fragment
    o0 = wmma_bf16(pf, load_frag_bf16(vp, 32,  0 + lm, 0, hi), o0);
    o1 = wmma_bf16(pf, load_frag_bf16(vp, 32, 16 + lm, 0, hi), o1);
    o2 = wmma_bf16(pf, load_frag_bf16(vp, 32, 32 + lm, 0, hi), o2);
    o3 = wmma_bf16(pf, load_frag_bf16(vp, 32, 48 + lm, 0, hi), o3);
  }

#pragma unroll
  for (int r = 0; r < 8; ++r) {
    float inv = 1.f / lrow[r];
    int row = q0 + r + 8 * hi;
    __bf16* dst = ob + (size_t)row * DIM + h * HDIM;
    dst[ 0 + lm] = (__bf16)(o0[r] * inv);
    dst[16 + lm] = (__bf16)(o1[r] * inv);
    dst[32 + lm] = (__bf16)(o2[r] * inv);
    dst[48 + lm] = (__bf16)(o3[r] * inv);
  }
}

// ------------- kernel 4: output expert projection --------------------------
__global__ __launch_bounds__(256) void wo_proj_kernel(
    const __bf16* __restrict__ ab, const __bf16* __restrict__ woT,
    const int* __restrict__ mod, float* __restrict__ obuf) {
  const int lane = threadIdx.x & 31, wave = threadIdx.x >> 5;
  const int hi = lane >> 4, lm = lane & 15;
  const int t0 = blockIdx.x * 16;
  const int c0 = (blockIdx.y * 8 + wave) * 16;
  const __bf16* w0 = woT;
  const __bf16* w1 = woT + (size_t)DIM * DIM;

  f32x8 acc0 = {}, acc1 = {};
  for (int k0 = 0; k0 < DIM; k0 += 32) {
    bf16x16 a  = load_frag_bf16(ab, DIM, t0 + lm, k0, hi);
    bf16x16 b0 = load_frag_bf16(w0, DIM, c0 + lm, k0, hi);
    bf16x16 b1 = load_frag_bf16(w1, DIM, c0 + lm, k0, hi);
    acc0 = wmma_bf16(a, b0, acc0);
    acc1 = wmma_bf16(a, b1, acc1);
  }
#pragma unroll
  for (int r = 0; r < 8; ++r) {
    int row = t0 + r + 8 * hi;
    obuf[(size_t)row * DIM + c0 + lm] = mod[row] ? acc1[r] : acc0[r];
  }
}

// ------------- kernel 5: per-token LayerNorm with modality gamma/beta ------
__global__ __launch_bounds__(256) void ln_kernel(
    const float* __restrict__ obuf, const int* __restrict__ mod,
    const float* __restrict__ ln_w, const float* __restrict__ ln_b,
    float* __restrict__ out) {
  __shared__ float red[2][8];
  const int n = blockIdx.x;
  const int lane = threadIdx.x & 31, wave = threadIdx.x >> 5;
  float s = 0.f, s2 = 0.f;
  for (int i = threadIdx.x; i < DIM; i += 256) {
    float v = obuf[(size_t)n * DIM + i];
    s += v; s2 += v * v;
  }
#pragma unroll
  for (int m = 1; m < 32; m <<= 1) {
    s += __shfl_xor(s, m, 32); s2 += __shfl_xor(s2, m, 32);
  }
  if (lane == 0) { red[0][wave] = s; red[1][wave] = s2; }
  __syncthreads();
  float ts = 0.f, ts2 = 0.f;
#pragma unroll
  for (int w = 0; w < 8; ++w) { ts += red[0][w]; ts2 += red[1][w]; }
  float mu = ts * (1.f / DIM);
  float var = ts2 * (1.f / DIM) - mu * mu;
  float rs = rsqrtf(var + 1e-5f);
  int m = mod[n];
  for (int i = threadIdx.x; i < DIM; i += 256) {
    float v = (obuf[(size_t)n * DIM + i] - mu) * rs;
    out[(size_t)n * DIM + i] = v * ln_w[(size_t)m * DIM + i] + ln_b[(size_t)m * DIM + i];
  }
}

// ---------------------------------------------------------------------------
extern "C" void kernel_launch(void* const* d_in, const int* in_sizes, int n_in,
                              void* d_out, int out_size, void* d_ws,
                              size_t ws_size, hipStream_t stream) {
  (void)in_sizes; (void)n_in; (void)out_size; (void)ws_size;
  const float*         x     = (const float*)d_in[0];
  const unsigned char* masks = (const unsigned char*)d_in[1];  // bool [M,N]
  const float*         wq    = (const float*)d_in[3];
  const float*         wk    = (const float*)d_in[4];
  const float*         wv    = (const float*)d_in[5];
  const float*         wo    = (const float*)d_in[6];
  const float*         ln_w  = (const float*)d_in[7];
  const float*         ln_b  = (const float*)d_in[8];
  float*               out   = (float*)d_out;

  // workspace carve-up (~44.3 MB)
  char* w = (char*)d_ws;
  int*    mod  = (int*)w;                                   // 8 KB
  __bf16* qb   = (__bf16*)(w + 8192);                       // 4 MB
  __bf16* kb   = qb + (size_t)N_TOK * DIM;                  // 4 MB
  __bf16* vT   = kb + (size_t)N_TOK * DIM;                  // 4 MB
  __bf16* ab   = vT + (size_t)N_TOK * DIM;                  // 4 MB
  float*  obuf = (float*)(ab + (size_t)N_TOK * DIM);        // 8 MB
  __bf16* wt   = (__bf16*)(obuf + (size_t)N_TOK * DIM);     // 12 MB (qkv^T)
  __bf16* woT  = wt + (size_t)3 * NMOD * DIM * DIM;         // 4 MB
  __bf16* xb   = woT + (size_t)NMOD * DIM * DIM;            // 4 MB

  mods_kernel<<<(N_TOK + 255) / 256, 256, 0, stream>>>(masks, mod);
  cvt_kernel<<<(N_TOK * DIM + 255) / 256, 256, 0, stream>>>(x, xb, N_TOK * DIM);

  const size_t WSZ = (size_t)DIM * DIM;
  int wgrid = (int)((WSZ + 255) / 256);
  for (int m = 0; m < NMOD; ++m) {
    wt_kernel<<<wgrid, 256, 0, stream>>>(wq + m * WSZ, wt + (0 * NMOD + m) * WSZ, DIM, DIM);
    wt_kernel<<<wgrid, 256, 0, stream>>>(wk + m * WSZ, wt + (1 * NMOD + m) * WSZ, DIM, DIM);
    wt_kernel<<<wgrid, 256, 0, stream>>>(wv + m * WSZ, wt + (2 * NMOD + m) * WSZ, DIM, DIM);
    wt_kernel<<<wgrid, 256, 0, stream>>>(wo + m * WSZ, woT + m * WSZ, DIM, DIM);
  }

  qkv_proj_kernel<<<dim3(N_TOK / 16, 24), 256, 0, stream>>>(xb, wt, mod, qb, kb, vT);
  attn_kernel<<<dim3(NHEAD, N_TOK / 16 / 8), 256, 0, stream>>>(qb, kb, vT, ab);
  wo_proj_kernel<<<dim3(N_TOK / 16, DIM / 16 / 8), 256, 0, stream>>>(ab, woT, mod, obuf);
  ln_kernel<<<N_TOK, 256, 0, stream>>>(obuf, mod, ln_w, ln_b, out);
}
